// Fedotp_50749333570187
// MI455X (gfx1250) — compile-verified, matrix operand order checked by reading, not run on hardware
//
#include <hip/hip_runtime.h>

typedef _Float16 v8h  __attribute__((ext_vector_type(8)));
typedef _Float16 v16h __attribute__((ext_vector_type(16)));
typedef float    v8f  __attribute__((ext_vector_type(8)));

#define M_ROWS   196
#define B_IMG    256
#define D_DIM    768
#define N_TXT    2
#define NCLS     200
#define IMG_ROWS (M_ROWS * B_IMG)          // 50176
#define TXT_ROWS (N_TXT * NCLS)            // 400
#define TOT_ROWS (IMG_ROWS + TXT_ROWS)     // 50576

#define NTILES   5                          // N-tiles (16 cols each) per workgroup
#define CT       8                          // classes per N-tile
#define MT       13                         // 16-row M tiles (208 >= 196)
#define LDSC     16

// ---------------------------------------------------------------------------
// Kernel 1: L2-normalize each 768-row of image & text features, emit f16.
// One wave32 per row; 24 elements per lane; butterfly reduce for ||x||^2.
// ---------------------------------------------------------------------------
__global__ __launch_bounds__(256) void fedotp_normalize(
    const float* __restrict__ img, const float* __restrict__ txt,
    _Float16* __restrict__ dst) {
  const int row  = (int)((blockIdx.x * blockDim.x + threadIdx.x) >> 5);
  const int lane = (int)(threadIdx.x & 31);
  if (row >= TOT_ROWS) return;
  const float* src = (row < IMG_ROWS) ? (img + (size_t)row * D_DIM)
                                      : (txt + (size_t)(row - IMG_ROWS) * D_DIM);
  _Float16* out = dst + (size_t)row * D_DIM;

  float x[D_DIM / 32];
  float ss = 0.0f;
#pragma unroll
  for (int i = 0; i < D_DIM / 32; ++i) {
    x[i] = src[lane + i * 32];
    ss += x[i] * x[i];
  }
#pragma unroll
  for (int off = 16; off >= 1; off >>= 1) ss += __shfl_xor(ss, off, 32);
  const float rn = __builtin_amdgcn_rsqf(ss);
#pragma unroll
  for (int i = 0; i < D_DIM / 32; ++i) out[lane + i * 32] = (_Float16)(x[i] * rn);
}

// ---------------------------------------------------------------------------
// Kernel 2: per (b, 40-class tile): WMMA GEMM with 5 accumulator N-tiles
// (one A fragment feeds 5 wmma per k-step -> 5x less L2 A-traffic), then the
// 5 tiles are drained through a single 13 KB LDS buffer, each followed by 8
// independent entropic-OT (Sinkhorn) problems on waves 0..7.
// ---------------------------------------------------------------------------
__global__ __launch_bounds__(MT * 32) void fedotp_main(
    const _Float16* __restrict__ nrm,      // [TOT_ROWS, 768] f16 (img then txt)
    const float* __restrict__ logit_scale,
    float* __restrict__ out) {
  __shared__ float sim[MT * 16 * LDSC];    // 208 x 16 f32, 13 KB (reused 5x)

  const int b      = (int)blockIdx.y;
  const int c0     = (int)blockIdx.x * (NTILES * CT);   // 40 classes / WG
  const int wave   = (int)(threadIdx.x >> 5);
  const int lane   = (int)(threadIdx.x & 31);
  const int laneHi = lane >> 4;            // 0 | 1
  const int lane15 = lane & 15;

  // ------------------- Phase 1: 5 sim tiles via v_wmma_f32_16x16x32_f16 ---
  v8f acc[NTILES];
#pragma unroll
  for (int t = 0; t < NTILES; ++t) acc[t] = (v8f){};

  {
    int m = wave * 16 + lane15;            // A row this lane feeds
    if (m > M_ROWS - 1) m = M_ROWS - 1;    // clamp padded rows (never read back)
    const _Float16* arow = nrm + ((size_t)m * B_IMG + b) * D_DIM;

    const int j  = lane15;                 // column within tile = c_local*2 + n
    const int n  = j & 1;
    const int cl = j >> 1;
    // tile t's column block starts CT classes further: + t*CT*D_DIM halves
    const _Float16* brow0 =
        nrm + (size_t)(IMG_ROWS + n * NCLS + c0 + cl) * D_DIM;

    const int ko = laneHi * 8;             // ISA 16-bit A/B lane K split

#pragma unroll 2
    for (int k = 0; k < D_DIM; k += 32) {
      v8h a0 = *(const v8h*)(arow + k + ko);        // K = k+ko .. +7
      v8h a1 = *(const v8h*)(arow + k + ko + 16);   // K = k+ko+16 .. +23
      v16h av = __builtin_shufflevector(a0, a1, 0, 1, 2, 3, 4, 5, 6, 7,
                                        8, 9, 10, 11, 12, 13, 14, 15);
#pragma unroll
      for (int t = 0; t < NTILES; ++t) {
        const _Float16* brow = brow0 + (size_t)t * CT * D_DIM;
        v8h b0 = *(const v8h*)(brow + k + ko);
        v8h b1 = *(const v8h*)(brow + k + ko + 16);
        v16h bv = __builtin_shufflevector(b0, b1, 0, 1, 2, 3, 4, 5, 6, 7,
                                          8, 9, 10, 11, 12, 13, 14, 15);
        acc[t] = __builtin_amdgcn_wmma_f32_16x16x32_f16(
            false, av, false, bv, (short)0, acc[t], false, false);
      }
    }
  }

  // ------------------- Phase 2: drain 5 tiles through LDS + Sinkhorn ------
  const int rbase = wave * 16 + laneHi * 8;   // f32 C/D layout: VGPR r -> M=r+8*laneHi
#pragma unroll
  for (int t = 0; t < NTILES; ++t) {
    __syncthreads();                       // previous tile's readers done
#pragma unroll
    for (int r = 0; r < 8; ++r)
      sim[(rbase + r) * LDSC + lane15] = acc[t][r];
    __syncthreads();                       // tile visible to all waves

    if (wave < CT) {
      const int j0 = wave * 2, j1 = j0 + 1;
      float K0[7], K1[7], S0[7], S1[7], U[7];
#pragma unroll
      for (int i = 0; i < 7; ++i) {
        const int m  = lane + 32 * i;
        const bool v = (m < M_ROWS);
        const int mc = v ? m : (M_ROWS - 1);
        const float s0 = sim[mc * LDSC + j0];
        const float s1 = sim[mc * LDSC + j1];
        S0[i] = s0; S1[i] = s1;
        K0[i] = v ? __expf((s0 - 1.0f) * 10.0f) : 0.0f;   // 1/eps = 10
        K1[i] = v ? __expf((s1 - 1.0f) * 10.0f) : 0.0f;
        U[i] = 1.0f;
      }

      float v0 = 1.0f, v1 = 1.0f;
      for (int it = 0; it < 100; ++it) {
        float a0 = 0.0f, a1 = 0.0f;
#pragma unroll
        for (int i = 0; i < 7; ++i) {
          // u = min(1 / (m * (K v)), 1);  a = 1/m uniform -> Kp = m*K
          const float tt = 196.0f * (K0[i] * v0 + K1[i] * v1);
          const float u  = fminf(__builtin_amdgcn_rcpf(tt), 1.0f);
          U[i] = u;
          a0 += K0[i] * u;
          a1 += K1[i] * u;
        }
#pragma unroll
        for (int off = 16; off >= 1; off >>= 1) {
          a0 += __shfl_xor(a0, off, 32);
          a1 += __shfl_xor(a1, off, 32);
        }
        // v = dy / (Kq u), Kq = K^T / 0.4  ->  v = 0.4 / (K^T u)
        const float v0n = 0.4f * __builtin_amdgcn_rcpf(a0);
        const float v1n = 0.4f * __builtin_amdgcn_rcpf(a1);
        const float err = 0.5f * (fabsf(v0n - v0) + fabsf(v1n - v1));
        v0 = v0n; v1 = v1n;
        if (err < 1e-3f) break;            // uniform across wave (identical sums)
      }

      // sim_op = sum_{m,n} u[m] * K[m,n] * v[n] * sim[m,n]
      float p = 0.0f;
#pragma unroll
      for (int i = 0; i < 7; ++i)
        p += U[i] * (K0[i] * v0 * S0[i] + K1[i] * v1 * S1[i]);
#pragma unroll
      for (int off = 16; off >= 1; off >>= 1) p += __shfl_xor(p, off, 32);

      if (lane == 0)
        out[(size_t)b * NCLS + c0 + t * CT + wave] =
            __expf(logit_scale[0]) * p;
    }
  }
}

// ---------------------------------------------------------------------------
extern "C" void kernel_launch(void* const* d_in, const int* in_sizes, int n_in,
                              void* d_out, int out_size, void* d_ws,
                              size_t ws_size, hipStream_t stream) {
  const float* img = (const float*)d_in[0];   // [196,256,768] f32
  const float* txt = (const float*)d_in[1];   // [2,200,768]   f32
  const float* ls  = (const float*)d_in[2];   // scalar
  float* out       = (float*)d_out;           // [256,200]     f32
  _Float16* nrm    = (_Float16*)d_ws;         // (50176+400)*768 f16 ~ 74 MB

  const int waves  = TOT_ROWS;                // one wave per row
  const int blocks = (waves + 7) / 8;         // 8 waves / 256-thread block
  fedotp_normalize<<<blocks, 256, 0, stream>>>(img, txt, nrm);

  dim3 grid(NCLS / (NTILES * CT), B_IMG);     // (5, 256)
  fedotp_main<<<grid, MT * 32, 0, stream>>>(nrm, ls, out);
}
